// GraphProcessor_80968723464404
// MI455X (gfx1250) — matrix operand order repaired and checked by reading, not compiled, not run on hardware
//
#include <hip/hip_runtime.h>
#include <hip/hip_bf16.h>

typedef _Float16 f16;
typedef __attribute__((ext_vector_type(16))) _Float16 v16h;
typedef __attribute__((ext_vector_type(8)))  float    v8f;

#define SDIM 4
#define BDIM 4
#define CDIM 64
#define HWDIM 1024
#define SB 16
#define SSB 64

static __device__ __forceinline__ v8f wmma_f16(v16h a, v16h b, v8f c) {
  // D = A(16x32 f16) * B(32x16 f16) + C(16x16 f32)
  return __builtin_amdgcn_wmma_f32_16x16x32_f16(false, a, false, b, (short)0, c, false, false);
}

// ---------------------------------------------------------------------------
// Implicit-GEMM conv, fully constant-specialized.
//   X : f16 pixel-major [nImg][1024][CIN]  (32x32 images)
//   Wt: f16 [KS*KS][COUT][CIN]
// Each wave computes 64 couts x 16 pixels (4 WMMA M-tiles sharing one B-frag).
// outTrans=1 -> output pixel-major [img][p][COUT], else channel-major.
// ---------------------------------------------------------------------------
template <int CIN, int COUT, int KS, int PAD>
__global__ void conv_wmma(const f16* __restrict__ X, const f16* __restrict__ Wt,
                          const float* __restrict__ bias,
                          float* __restrict__ outF, f16* __restrict__ outH, int outTrans,
                          int nImg)
{
  constexpr int MQ  = COUT >> 6;          // 64-cout quads
  constexpr int TPI = MQ * 64;            // power of two (64 or 128)
  constexpr int KQ  = CIN >> 5;

  int wave = blockIdx.x * (blockDim.x >> 5) + (threadIdx.x >> 5);
  int lane = threadIdx.x & 31;
  int l = lane & 15, hf = lane >> 4;
  int img = wave / TPI;                   // compile-time pow2 -> shift
  int rem = wave & (TPI - 1);
  int mqi = rem >> 6;
  int nt  = rem & 63;
  if (img >= nImg) return;
  int m0 = mqi << 6;
  int p0 = nt << 4;
  int y  = p0 >> 5, xb = p0 & 31;
  int ka0 = hf ? 16 : 0, ka1 = hf ? 24 : 8;

  v8f acc[4] = {};
  int t = 0;
  for (int dy = -PAD; dy <= PAD; ++dy) {
    for (int dx = -PAD; dx <= PAD; ++dx, ++t) {
      int yy = y + dy;
      int xx = xb + l + dx;
      bool ok = (yy >= 0) && (yy < 32) && (xx >= 0) && (xx < 32);
      const f16* xp = nullptr;
      if (ok) {
        xp = X + ((size_t)img * HWDIM + (yy * 32 + xx)) * CIN;
        __builtin_prefetch(xp + 32 * CIN, 0, 3);   // next tap row, near-cache
      }
      #pragma unroll
      for (int kk = 0; kk < KQ; ++kk) {
        int kb = kk << 5;
        v16h b;                            // B: 32x16, k=ci, n=pixel (contiguous)
        if (ok) {
          const f16* bp = xp + kb + hf * 16;
          #pragma unroll
          for (int j = 0; j < 16; ++j) b[j] = bp[j];
        } else {
          #pragma unroll
          for (int j = 0; j < 16; ++j) b[j] = (f16)0.0f;
        }
        #pragma unroll
        for (int mi = 0; mi < 4; ++mi) {   // A: 16x32 of W_tap, row m = l
          const f16* wrow = Wt + ((size_t)t * COUT + m0 + mi * 16 + l) * CIN + kb;
          v16h a;
          #pragma unroll
          for (int j = 0; j < 8; ++j) { a[j] = wrow[ka0 + j]; a[j + 8] = wrow[ka1 + j]; }
          acc[mi] = wmma_f16(a, b, acc[mi]);
        }
      }
    }
  }
  int p = p0 + l;
  #pragma unroll
  for (int mi = 0; mi < 4; ++mi) {
    #pragma unroll
    for (int r = 0; r < 8; ++r) {
      int co = m0 + mi * 16 + hf * 8 + r;
      float v = acc[mi][r] + (bias ? bias[co] : 0.0f);
      size_t o = outTrans ? ((size_t)img * HWDIM + p) * COUT + co
                          : ((size_t)img * COUT + co) * HWDIM + p;
      if (outF) outF[o] = v;
      if (outH) outH[o] = (f16)v;
    }
  }
}

// ---------------------------------------------------------------------------
// Flash attention.
//   Q, K: f16 pixel-major [1024][64] per head   (contiguous fragment loads)
//   V   : f16 channel-major [64][1024] per head (PV B-frag contiguous in n)
// logits[m,n] = sum_c Q[m,c]*K[n,c]; softmax over n; O[c,m] = sum_n P[m,n]*V[c,n].
// Two-pass (row-max, then exp + P*V). One wave = 16 rows of one head.
// ---------------------------------------------------------------------------
__global__ void flash_attn(const f16* __restrict__ Qb, const f16* __restrict__ Kb,
                           const f16* __restrict__ Vb,
                           float* __restrict__ outF, f16* __restrict__ outH, int outTrans,
                           const float* __restrict__ resid,
                           const float* __restrict__ alphaPtr, int mode)
{
  __shared__ float ldsP[8][16 * 32];   // per-wave P-tile staging (D -> A layout transpose)
  int wave = blockIdx.x * (blockDim.x >> 5) + (threadIdx.x >> 5);
  int lane = threadIdx.x & 31;
  int l = lane & 15, hf = lane >> 4;
  int mt = wave & 63;
  int head = wave >> 6;
  size_t qoff, koff, voff;
  if (mode == 0) {                     // intra: head = s*B+b
    qoff = koff = voff = (size_t)head * (CDIM * HWDIM);
  } else {                             // inter: head = ((i*S+j)*B+b)
    int i = head >> 4, j = (head >> 2) & 3, b = head & 3;
    qoff = (size_t)(i * BDIM + b) * (CDIM * HWDIM);
    koff = (size_t)(j * BDIM + b) * (CDIM * HWDIM);
    voff = koff;
  }
  size_t ooff = (size_t)head * (CDIM * HWDIM);
  const f16* Q = Qb + qoff;
  const f16* K = Kb + koff;
  const f16* V = Vb + voff;
  int m0 = mt << 4;
  int ka0 = hf ? 16 : 0, ka1 = hf ? 24 : 8;

  // Preload Q A-fragments once (reused by every n-tile in both passes).
  v16h aq[2];
  {
    const f16* qrow = Q + (size_t)(m0 + l) * CDIM;
    #pragma unroll
    for (int kk = 0; kk < 2; ++kk) {
      #pragma unroll
      for (int j = 0; j < 8; ++j) {
        aq[kk][j]     = qrow[kk * 32 + ka0 + j];
        aq[kk][j + 8] = qrow[kk * 32 + ka1 + j];
      }
    }
  }

  // pass 1: row maxima
  float mx[8];
  #pragma unroll
  for (int r = 0; r < 8; ++r) mx[r] = -1e30f;
  for (int nt = 0; nt < 64; ++nt) {
    __builtin_prefetch(K + (size_t)((((nt + 4) & 63) << 4) + l) * CDIM, 0, 3);
    const f16* krow = K + (size_t)((nt << 4) + l) * CDIM;
    v8f s = {};
    #pragma unroll
    for (int kk = 0; kk < 2; ++kk) {
      v16h b;
      #pragma unroll
      for (int j = 0; j < 16; ++j) b[j] = krow[kk * 32 + hf * 16 + j];
      s = wmma_f16(aq[kk], b, s);
    }
    #pragma unroll
    for (int r = 0; r < 8; ++r) mx[r] = fmaxf(mx[r], s[r]);
  }
  #pragma unroll
  for (int r = 0; r < 8; ++r)
    for (int o = 1; o < 16; o <<= 1)
      mx[r] = fmaxf(mx[r], __shfl_xor(mx[r], o, 32));

  // pass 2: exp, row sums, P*V
  v8f oa[4] = {};
  float rs[8];
  #pragma unroll
  for (int r = 0; r < 8; ++r) rs[r] = 0.0f;
  float* P = ldsP[threadIdx.x >> 5];
  for (int nc = 0; nc < 32; ++nc) {
    int n0 = nc << 5;
    v8f s0 = {}, s1 = {};
    {
      const f16* krow0 = K + (size_t)(n0 + l) * CDIM;
      const f16* krow1 = K + (size_t)(n0 + 16 + l) * CDIM;
      #pragma unroll
      for (int kk = 0; kk < 2; ++kk) {
        v16h b0, b1;
        #pragma unroll
        for (int j = 0; j < 16; ++j) {
          b0[j] = krow0[kk * 32 + hf * 16 + j];
          b1[j] = krow1[kk * 32 + hf * 16 + j];
        }
        s0 = wmma_f16(aq[kk], b0, s0);
        s1 = wmma_f16(aq[kk], b1, s1);
      }
    }
    #pragma unroll
    for (int r = 0; r < 8; ++r) {
      float p0 = __expf(s0[r] - mx[r]);
      float p1 = __expf(s1[r] - mx[r]);
      rs[r] += p0 + p1;
      int row = hf * 8 + r;
      P[row * 32 + l]      = p0;
      P[row * 32 + 16 + l] = p1;
    }
    __asm__ volatile("" ::: "memory"); // compiler fence; wave-local LDS ops are in-order
    v16h a;                            // A: P 16x32, row m = l
    #pragma unroll
    for (int j = 0; j < 8; ++j) {
      a[j]     = (f16)P[l * 32 + ka0 + j];
      a[j + 8] = (f16)P[l * 32 + ka1 + j];
    }
    __asm__ volatile("" ::: "memory");
    #pragma unroll
    for (int ct = 0; ct < 4; ++ct) {   // B: V 32x16 (k=n local, n=c), contiguous in n
      const f16* vrow = V + (size_t)(ct * 16 + l) * HWDIM + n0 + hf * 16;
      v16h b;
      #pragma unroll
      for (int j = 0; j < 16; ++j) b[j] = vrow[j];
      oa[ct] = wmma_f16(a, b, oa[ct]);
    }
  }
  #pragma unroll
  for (int r = 0; r < 8; ++r)
    for (int o = 1; o < 16; o <<= 1)
      rs[r] += __shfl_xor(rs[r], o, 32);

  float al = alphaPtr ? alphaPtr[0] : 1.0f;
  #pragma unroll
  for (int ct = 0; ct < 4; ++ct) {
    #pragma unroll
    for (int r = 0; r < 8; ++r) {
      int c = ct * 16 + l;
      int m = m0 + hf * 8 + r;
      float v = oa[ct][r] / rs[r];
      size_t o = outTrans ? ooff + (size_t)m * CDIM + c
                          : ooff + (size_t)c * HWDIM + m;
      if (resid) v = al * v + resid[o];
      if (outF) outF[o] = v;
      if (outH) outH[o] = (f16)v;
    }
  }
}

// ---------------------------------------------------------------------------
// Pointwise / reduction helper kernels
// ---------------------------------------------------------------------------
// grid: (O*I/256, taps); src OIHW [o][i][t] -> dst [t][o][i]. iShift = log2(I).
__global__ void prep_weights(const float* __restrict__ src, f16* __restrict__ dst,
                             int O, int iShift, int taps)
{
  int t  = blockIdx.y;
  int oi = blockIdx.x * blockDim.x + threadIdx.x;
  int i  = oi & ((1 << iShift) - 1);
  int o  = oi >> iShift;
  if (o >= O) return;
  dst[(size_t)t * (O << iShift) + oi] = (f16)src[(((size_t)o << iShift) + i) * taps + t];
}

__global__ void copy_f32(const float* __restrict__ src, float* __restrict__ dst, int n)
{
  int idx = blockIdx.x * blockDim.x + threadIdx.x;
  if (idx < n) dst[idx] = src[idx];
}

// h (f32 channel-major) -> hh16 (channel-major), hT16 (pixel-major),
// cat1 upper 64 channels (pixel-major [sb][p][128])
__global__ void cvt_h(const float* __restrict__ h, f16* __restrict__ hh,
                      f16* __restrict__ hT, f16* __restrict__ cat1)
{
  int idx = blockIdx.x * blockDim.x + threadIdx.x;
  if (idx >= SB * CDIM * HWDIM) return;
  float v = h[idx];
  hh[idx] = (f16)v;
  int sb = idx >> 16;
  int c  = (idx >> 10) & 63;
  int p  = idx & 1023;
  hT[((size_t)sb * HWDIM + p) * CDIM + c] = (f16)v;
  cat1[((size_t)sb * HWDIM + p) * 128 + 64 + c] = (f16)v;
}

__global__ void gate_kernel(const float* __restrict__ gconv, float* __restrict__ gate)
{
  int w = blockIdx.x * (blockDim.x >> 5) + (threadIdx.x >> 5); // one wave per (img,c)
  int lane = threadIdx.x & 31;
  const float* src = gconv + (size_t)w * HWDIM;
  float s = 0.f;
  for (int i = lane; i < HWDIM; i += 32) s += src[i];
  #pragma unroll
  for (int o = 1; o < 32; o <<= 1) s += __shfl_xor(s, o, 32);
  if (lane == 0) gate[w] = 1.f / (1.f + __expf(-s * (1.0f / 1024.0f)));
}

// BN + masked j-sum + agg -> cat1 lower 64 channels (pixel-major)
__global__ void inter_agg(const float* __restrict__ intra, const f16* __restrict__ mji,
                          const float* __restrict__ gate,
                          const float* __restrict__ gamma, const float* __restrict__ beta,
                          const float* __restrict__ mean, const float* __restrict__ var,
                          const float* __restrict__ intraW, const float* __restrict__ interW,
                          f16* __restrict__ cat1)
{
  int idx = blockIdx.x * blockDim.x + threadIdx.x;
  if (idx >= SB * CDIM * HWDIM) return;
  int sb = idx >> 16;
  int s = sb >> 2, b = sb & 3;
  int c = (idx >> 10) & 63;
  int p = idx & 1023;
  float rv = rsqrtf(var[c] + 1e-5f);
  float mu = mean[c], ga = gamma[c], be = beta[c];
  float acc = 0.f;
  #pragma unroll
  for (int j = 0; j < SDIM; ++j) {
    if (j == s) continue;
    int img = (s * SDIM + j) * BDIM + b;
    float g = gate[img * CDIM + c];
    float v = (float)mji[((size_t)img * HWDIM + p) * CDIM + c];   // mji is pixel-major
    acc += (g * v - mu) * rv * ga + be;
  }
  float agg = intraW[0] * intra[idx] + interW[0] * acc;
  cat1[((size_t)sb * HWDIM + p) * 128 + c] = (f16)agg;
}

// zr_raw: f32 channel-major [sb][128][1024]; writes z (f32) and cat1 upper = r*h
__global__ void gru_zr_post(const float* __restrict__ zr, const float* __restrict__ h,
                            float* __restrict__ zbuf, f16* __restrict__ cat1)
{
  int idx = blockIdx.x * blockDim.x + threadIdx.x;
  if (idx >= SB * CDIM * HWDIM) return;
  int sb = idx >> 16;
  int c  = (idx >> 10) & 63;
  int p  = idx & 1023;
  size_t zi = (size_t)sb * (128 * 1024) + (size_t)c * 1024 + p;
  size_t ri = zi + 65536;
  float zv = 1.f / (1.f + __expf(-zr[zi]));
  float rv = 1.f / (1.f + __expf(-zr[ri]));
  zbuf[idx] = zv;
  cat1[((size_t)sb * HWDIM + p) * 128 + 64 + c] = (f16)(rv * h[idx]);
}

__global__ void gru_final(const float* __restrict__ hh_raw, const float* __restrict__ zbuf,
                          float* __restrict__ h)
{
  int idx = blockIdx.x * blockDim.x + threadIdx.x;
  if (idx >= SB * CDIM * HWDIM) return;
  float hg = h[idx];
  float z = zbuf[idx];
  float hh = tanhf(hh_raw[idx]);
  h[idx] = (1.f - z) * hg + z * hh + hg;
}

// ---------------------------------------------------------------------------
extern "C" void kernel_launch(void* const* d_in, const int* in_sizes, int n_in,
                              void* d_out, int out_size, void* d_ws, size_t ws_size,
                              hipStream_t stream)
{
  const float* x      = (const float*)d_in[0];
  const float* Wf_w   = (const float*)d_in[1];
  const float* Wf_b   = (const float*)d_in[2];
  const float* Wh_w   = (const float*)d_in[3];
  const float* Wh_b   = (const float*)d_in[4];
  const float* Wl_w   = (const float*)d_in[5];
  const float* Wl_b   = (const float*)d_in[6];
  const float* alpha  = (const float*)d_in[7];
  const float* Wc     = (const float*)d_in[8];
  const float* Wg_w   = (const float*)d_in[9];
  const float* Wg_b   = (const float*)d_in[10];
  const float* bn_g   = (const float*)d_in[11];
  const float* bn_b   = (const float*)d_in[12];
  const float* bn_m   = (const float*)d_in[13];
  const float* bn_v   = (const float*)d_in[14];
  const float* zr_w   = (const float*)d_in[15];
  const float* zr_b   = (const float*)d_in[16];
  const float* hw_w   = (const float*)d_in[17];
  const float* hw_b   = (const float*)d_in[18];
  const float* intraW = (const float*)d_in[19];
  const float* interW = (const float*)d_in[20];
  (void)in_sizes; (void)n_in; (void)out_size; (void)ws_size;

  char* ws = (char*)d_ws;
  size_t cur = 0;
  auto alloc = [&](size_t bytes) -> void* {
    void* p = ws + cur;
    cur += (bytes + 255) & ~(size_t)255;
    return p;
  };
  const size_t NE = (size_t)SB * CDIM * HWDIM;   // 1,048,576
  float* h      = (float*)alloc(NE * 4);
  f16*   hh16   = (f16*)  alloc(NE * 2);          // channel-major h
  f16*   hT16   = (f16*)  alloc(NE * 2);          // pixel-major h
  f16*   ft16   = (f16*)  alloc(NE * 2);          // pixel-major
  f16*   ht16   = (f16*)  alloc(NE * 2);          // pixel-major
  f16*   lt16   = (f16*)  alloc(NE * 2);          // channel-major
  f16*   yt16   = (f16*)  alloc(NE * 2);          // pixel-major
  float* intra  = (float*)alloc(NE * 4);          // channel-major
  f16*   mji16  = (f16*)  alloc((size_t)SSB * CDIM * HWDIM * 2);  // pixel-major
  float* gconv  = (float*)alloc((size_t)SSB * CDIM * HWDIM * 4);  // channel-major
  float* gate   = (float*)alloc((size_t)SSB * CDIM * 4);
  f16*   cat1   = (f16*)  alloc((size_t)SB * 128 * HWDIM * 2);    // pixel-major [sb][p][128]
  float* zr_raw = (float*)alloc((size_t)SB * 128 * HWDIM * 4);    // channel-major
  float* zbuf   = (float*)alloc(NE * 4);
  float* hh_raw = (float*)alloc(NE * 4);          // channel-major
  f16*   wf     = (f16*)alloc((size_t)9 * 64 * 64 * 2);
  f16*   whw    = (f16*)alloc((size_t)9 * 64 * 64 * 2);
  f16*   wl     = (f16*)alloc((size_t)9 * 64 * 64 * 2);
  f16*   wg     = (f16*)alloc((size_t)9 * 64 * 64 * 2);
  f16*   wc     = (f16*)alloc((size_t)64 * 64 * 2);
  f16*   wzr    = (f16*)alloc((size_t)25 * 128 * 128 * 2);
  f16*   whh    = (f16*)alloc((size_t)25 * 64 * 128 * 2);

  const int T = 256;
  auto g1 = [](size_t n, int t) { return (unsigned)((n + (size_t)t - 1) / (size_t)t); };

  prep_weights<<<dim3(g1(64 * 64, T), 9),  T, 0, stream>>>(Wf_w, wf, 64, 6, 9);
  prep_weights<<<dim3(g1(64 * 64, T), 9),  T, 0, stream>>>(Wh_w, whw, 64, 6, 9);
  prep_weights<<<dim3(g1(64 * 64, T), 9),  T, 0, stream>>>(Wl_w, wl, 64, 6, 9);
  prep_weights<<<dim3(g1(64 * 64, T), 9),  T, 0, stream>>>(Wg_w, wg, 64, 6, 9);
  prep_weights<<<dim3(g1(64 * 64, T), 1),  T, 0, stream>>>(Wc, wc, 64, 6, 1);
  prep_weights<<<dim3(g1(128 * 128, T), 25), T, 0, stream>>>(zr_w, wzr, 128, 7, 25);
  prep_weights<<<dim3(g1(64 * 128, T), 25),  T, 0, stream>>>(hw_w, whh, 64, 7, 25);
  copy_f32<<<g1(NE, T), T, 0, stream>>>(x, h, (int)NE);

  for (int it = 0; it < 3; ++it) {
    cvt_h<<<g1(NE, T), T, 0, stream>>>(h, hh16, hT16, cat1);
    // feature convs (implicit GEMM); ft/ht/yt transposed out for attention Q/K
    conv_wmma<64, 64, 3, 1><<<128, T, 0, stream>>>(hT16, wf, Wf_b, nullptr, ft16, 1, 16);
    conv_wmma<64, 64, 3, 1><<<128, T, 0, stream>>>(hT16, whw, Wh_b, nullptr, ht16, 1, 16);
    conv_wmma<64, 64, 3, 1><<<128, T, 0, stream>>>(hT16, wl, Wl_b, nullptr, lt16, 0, 16);
    // yt = Wc @ h == 1x1 conv (pixel-major out, used as inter K)
    conv_wmma<64, 64, 1, 0><<<128, T, 0, stream>>>(hT16, wc, nullptr, nullptr, yt16, 1, 16);
    // intra attention, fused  intra = alpha*ctx + h (channel-major f32 out)
    flash_attn<<<128, T, 0, stream>>>(ft16, ht16, lt16, intra, nullptr, 0, h, alpha, 0);
    // inter attention -> mji (pixel-major f16)
    flash_attn<<<512, T, 0, stream>>>(hT16, yt16, hh16, nullptr, mji16, 1, nullptr, nullptr, 1);
    // gating conv on 64 pair-images (channel-major f32 out for the mean reduce)
    conv_wmma<64, 64, 3, 1><<<512, T, 0, stream>>>(mji16, wg, Wg_b, gconv, nullptr, 0, 64);
    gate_kernel<<<512, T, 0, stream>>>(gconv, gate);
    // BN + masked j-sum + agg -> cat1 lower channels (pixel-major)
    inter_agg<<<g1(NE, T), T, 0, stream>>>(intra, mji16, gate, bn_g, bn_b, bn_m, bn_v,
                                           intraW, interW, cat1);
    // GRU: 5x5 convs with Cin=128 via WMMA
    conv_wmma<128, 128, 5, 2><<<256, T, 0, stream>>>(cat1, wzr, zr_b, zr_raw, nullptr, 0, 16);
    gru_zr_post<<<g1(NE, T), T, 0, stream>>>(zr_raw, h, zbuf, cat1);
    conv_wmma<128, 64, 5, 2><<<128, T, 0, stream>>>(cat1, whh, hw_b, hh_raw, nullptr, 0, 16);
    gru_final<<<g1(NE, T), T, 0, stream>>>(hh_raw, zbuf, h);
  }
  copy_f32<<<g1(NE, T), T, 0, stream>>>(h, (float*)d_out, (int)NE);
}
